// SSMBlock_38208029065305
// MI455X (gfx1250) — matrix operand order repaired and checked by reading, not compiled, not run on hardware
//
#include <hip/hip_runtime.h>
#include <hip/hip_bf16.h>

#define INPUT_DIM 1024
#define STATE_DIM 16
#define CONV_K    4
#define INNER     2048
#define BATCH     2
#define SEQ       2048
#define ROWS      (BATCH * SEQ)   /* 4096 */

typedef __attribute__((ext_vector_type(8)))  __bf16 bf16x8;
typedef __attribute__((ext_vector_type(16))) __bf16 bf16x16;
typedef __attribute__((ext_vector_type(8)))  float  f32x8;
typedef __attribute__((ext_vector_type(4)))  int    v4i;

typedef __attribute__((address_space(1))) v4i as1_v4i;
typedef __attribute__((address_space(3))) v4i as3_v4i;

// ---------------------------------------------------------------------------
// Async global->LDS copy (CDNA5 GLOBAL_LOAD_ASYNC_TO_LDS_B128, ASYNCcnt
// tracked). Signature probed from the round-2 diagnostic:
//   (int4 addrspace(1)* src, int4 addrspace(3)* dst, imm offset, imm cpol)
// Guarded so a toolchain without the builtin still compiles (sync fallback).
// ---------------------------------------------------------------------------
#if __has_builtin(__builtin_amdgcn_global_load_async_to_lds_b128)
#define HAVE_ASYNC_LDS 1
#define ASYNC_COPY16(gsrc, ldst)                                              \
  __builtin_amdgcn_global_load_async_to_lds_b128(                             \
      (as1_v4i*)(uintptr_t)(gsrc),                                            \
      (as3_v4i*)(unsigned)(uintptr_t)(ldst), 0, 0)
#if __has_builtin(__builtin_amdgcn_s_wait_asynccnt)
#define ASYNC_WAIT() __builtin_amdgcn_s_wait_asynccnt(0)
#else
#define ASYNC_WAIT() asm volatile("s_wait_asynccnt 0x0" ::: "memory")
#endif
#else
#define HAVE_ASYNC_LDS 0
#define ASYNC_COPY16(gsrc, ldst)                                              \
  (*(bf16x8*)(ldst) = *(const bf16x8*)(gsrc))
#define ASYNC_WAIT() ((void)0)
#endif

static __device__ __forceinline__ __bf16 to_bf16(float f) {
  unsigned u = __builtin_bit_cast(unsigned, f);
  u += 0x7FFFu + ((u >> 16) & 1u);          // round-to-nearest-even
  unsigned short h = (unsigned short)(u >> 16);
  return __builtin_bit_cast(__bf16, h);
}

__global__ void f32_to_bf16_kernel(const float* __restrict__ in,
                                   __bf16* __restrict__ out, int n) {
  int i = blockIdx.x * blockDim.x + threadIdx.x;
  if (i < n) out[i] = to_bf16(in[i]);
}

// ---------------------------------------------------------------------------
// C(MxN) = A(MxK) * W(NxK)^T, bf16 in / fp32 out.
// Block = 4 waves = 128x128 macro-tile (2x2 waves, each a 64x64 tile of
// 4x4 WMMA 16x16x32 accumulators). K stepped by 32 with double-buffered
// 128x32 A/B tiles staged in LDS via async copies.
// ---------------------------------------------------------------------------
__global__ __launch_bounds__(128) void wmma_gemm_bf16(
    const __bf16* __restrict__ A,
    const __bf16* __restrict__ W,
    float* __restrict__ C,
    int M, int N, int K)
{
  __shared__ __bf16 As[2][128][32];   // 2 x 8KB
  __shared__ __bf16 Bs[2][128][32];   // 2 x 8KB

  const int tid  = threadIdx.x;
  const int lane = tid & 31;
  const int wave = tid >> 5;

  const int tilesN = N >> 7;
  const int btm = (blockIdx.x / tilesN) << 7;   // block M origin
  const int btn = (blockIdx.x % tilesN) << 7;   // block N origin
  const int wm  = (wave >> 1) << 6;             // wave M offset in block
  const int wn  = (wave & 1) << 6;              // wave N offset in block

  const int mh = lane & 15;   // row in 16-row fragment
  const int kg = lane >> 4;   // K-group selector per ISA layout

  // Stage one 128x32 A tile + 128x32 B tile into LDS buffer b at k-offset k0.
  // 512 chunks of 16B per tile; thread covers 4 chunks, 4 consecutive lanes
  // cover one 64B row (contiguous global addresses within a lane-quad).
  auto copy_tiles = [&](int b, int k0) {
#pragma unroll
    for (int r = 0; r < 4; ++r) {
      const int c    = r * 128 + tid;
      const int row  = c >> 2;
      const int colb = (c & 3) * 16;
      const char* ga = (const char*)A + ((size_t)(btm + row) * K + k0) * 2 + colb;
      ASYNC_COPY16(ga, (char*)(&As[b][0][0]) + c * 16);
      const char* gb = (const char*)W + ((size_t)(btn + row) * K + k0) * 2 + colb;
      ASYNC_COPY16(gb, (char*)(&Bs[b][0][0]) + c * 16);
    }
  };

  f32x8 acc[4][4];
#pragma unroll
  for (int i = 0; i < 4; ++i)
#pragma unroll
    for (int j = 0; j < 4; ++j)
      acc[i][j] = (f32x8){0.f,0.f,0.f,0.f,0.f,0.f,0.f,0.f};

  int buf = 0;
  copy_tiles(buf, 0);
  ASYNC_WAIT();
  __syncthreads();

  for (int k0 = 0; k0 < K; k0 += 32) {
    if (k0 + 32 < K) copy_tiles(buf ^ 1, k0 + 32);

    bf16x16 af[4], bfm[4];
    // A 16x32 fragment from LDS: lane holds K in [kg*8,kg*8+8) u [kg*8+16,+24)
#pragma unroll
    for (int i = 0; i < 4; ++i) {
      const __bf16* p = &As[buf][wm + i * 16 + mh][kg * 8];
      bf16x8 lo = *(const bf16x8*)p;
      bf16x8 hi = *(const bf16x8*)(p + 16);
      af[i] = __builtin_shufflevector(lo, hi,
                0,1,2,3,4,5,6,7,8,9,10,11,12,13,14,15);
    }
    // B 32x16 fragment from LDS: lane (col n=mh) holds K in [kg*8,kg*8+8) u [kg*8+16,+24)
#pragma unroll
    for (int j = 0; j < 4; ++j) {
      const __bf16* p = &Bs[buf][wn + j * 16 + mh][kg * 8];
      bf16x8 lo = *(const bf16x8*)p;
      bf16x8 hi = *(const bf16x8*)(p + 16);
      bfm[j] = __builtin_shufflevector(lo, hi,
                0,1,2,3,4,5,6,7,8,9,10,11,12,13,14,15);
    }
#pragma unroll
    for (int i = 0; i < 4; ++i)
#pragma unroll
      for (int j = 0; j < 4; ++j)
        acc[i][j] = __builtin_amdgcn_wmma_f32_16x16x32_bf16(
            /*neg_a=*/false, af[i], /*neg_b=*/false, bfm[j],
            /*c_mod=*/(short)0, acc[i][j],
            /*reuse_a=*/false, /*reuse_b=*/false);

    if (k0 + 32 < K) {
      ASYNC_WAIT();
      __syncthreads();
      buf ^= 1;
    }
  }

  // D layout: VGPR r -> (M = r + 8*kg, N = mh)
#pragma unroll
  for (int i = 0; i < 4; ++i)
#pragma unroll
    for (int j = 0; j < 4; ++j) {
      const int n = btn + wn + j * 16 + mh;
#pragma unroll
      for (int r = 0; r < 8; ++r) {
        const int m = btm + wm + i * 16 + kg * 8 + r;
        C[(size_t)m * N + n] = acc[i][j][r];
      }
    }
}

// ---------------------------------------------------------------------------
// Fused depthwise conv + SiLU + softplus + selective-scan + gating.
// One thread per (batch, channel); hidden state (16) lives in registers.
// ---------------------------------------------------------------------------
__global__ __launch_bounds__(256) void ssm_scan_kernel(
    const float* __restrict__ xr,      // (ROWS, 2*INNER): [:INNER]=x_ssm, [INNER:]=res
    const float* __restrict__ conv_w,  // (INNER, 1, 4)
    const float* __restrict__ conv_b,  // (INNER,)
    const float* __restrict__ A_log,   // (INNER, 16)
    const float* __restrict__ Dp,      // (INNER,)
    __bf16* __restrict__ ybuf)         // (ROWS, INNER) bf16
{
  const int tid = blockIdx.x * blockDim.x + threadIdx.x;
  if (tid >= BATCH * INNER) return;
  const int b = tid / INNER;
  const int c = tid % INNER;

  const float w0 = conv_w[c * 4 + 0];
  const float w1 = conv_w[c * 4 + 1];
  const float w2 = conv_w[c * 4 + 2];
  const float w3 = conv_w[c * 4 + 3];
  const float bias = conv_b[c];
  const float d = Dp[c];

  float Arow[STATE_DIM];
#pragma unroll
  for (int s = 0; s < STATE_DIM; ++s)
    Arow[s] = -__expf(A_log[c * STATE_DIM + s]);

  float h[STATE_DIM];
#pragma unroll
  for (int s = 0; s < STATE_DIM; ++s) h[s] = 0.f;

  float win0 = 0.f, win1 = 0.f, win2 = 0.f;  // x_ssm at t-3, t-2, t-1

  for (int t = 0; t < SEQ; ++t) {
    const size_t row = (size_t)(b * SEQ + t) * (2 * INNER);
    const float xs = xr[row + c];
    const float rs = xr[row + INNER + c];

    const float cv = w0 * win0 + w1 * win1 + w2 * win2 + w3 * xs + bias;
    const float xc = cv / (1.f + __expf(-cv));                       // SiLU
    const float delta = (xc > 20.f) ? xc : __logf(1.f + __expf(xc)); // softplus
    const float bx = xc * xc;

    float hsum = 0.f;
#pragma unroll
    for (int s = 0; s < STATE_DIM; ++s) {
      const float a = __expf(delta * Arow[s]);
      h[s] = a * h[s] + bx;
      hsum += h[s];
    }
    const float y = xc * hsum;
    const float sres = rs / (1.f + __expf(-rs));                     // SiLU(res)
    const float yf = y * sres + d * xc;

    ybuf[(size_t)(b * SEQ + t) * INNER + c] = to_bf16(yf);

    win0 = win1; win1 = win2; win2 = xs;
  }
}

extern "C" void kernel_launch(void* const* d_in, const int* in_sizes, int n_in,
                              void* d_out, int out_size, void* d_ws, size_t ws_size,
                              hipStream_t stream) {
  const float* x      = (const float*)d_in[0];
  const float* W_in   = (const float*)d_in[1];
  const float* conv_w = (const float*)d_in[2];
  const float* conv_b = (const float*)d_in[3];
  const float* A_log  = (const float*)d_in[4];
  const float* Dvec   = (const float*)d_in[5];
  const float* W_out  = (const float*)d_in[6];
  float* out = (float*)d_out;

  char* ws = (char*)d_ws;
  size_t off = 0;
  auto alloc = [&](size_t bytes) -> char* {
    char* p = ws + off;
    off += (bytes + 255) & ~(size_t)255;
    return p;
  };
  __bf16* xb  = (__bf16*)alloc((size_t)ROWS * INPUT_DIM * 2);        // x in bf16
  __bf16* wib = (__bf16*)alloc((size_t)(2 * INNER) * INPUT_DIM * 2); // W_in bf16
  float*  xr  = (float*) alloc((size_t)ROWS * (2 * INNER) * 4);      // GEMM1 out
  __bf16* yb  = (__bf16*)alloc((size_t)ROWS * INNER * 2);            // scan out bf16
  __bf16* wob = (__bf16*)alloc((size_t)INPUT_DIM * INNER * 2);       // W_out bf16

  {
    int n = ROWS * INPUT_DIM;
    f32_to_bf16_kernel<<<(n + 255) / 256, 256, 0, stream>>>(x, xb, n);
  }
  {
    int n = 2 * INNER * INPUT_DIM;
    f32_to_bf16_kernel<<<(n + 255) / 256, 256, 0, stream>>>(W_in, wib, n);
  }
  {
    int n = INPUT_DIM * INNER;
    f32_to_bf16_kernel<<<(n + 255) / 256, 256, 0, stream>>>(W_out, wob, n);
  }

  {
    // GEMM1: xr(4096x4096) = x(4096x1024) @ W_in(4096x1024)^T
    int M = ROWS, N = 2 * INNER, K = INPUT_DIM;
    int blocks = (M >> 7) * (N >> 7);           // 1024 blocks of 128 threads
    wmma_gemm_bf16<<<blocks, 128, 0, stream>>>(xb, wib, xr, M, N, K);
  }
  {
    int n = BATCH * INNER;                      // 4096 threads
    ssm_scan_kernel<<<(n + 255) / 256, 256, 0, stream>>>(
        xr, conv_w, conv_b, A_log, Dvec, yb);
  }
  {
    // GEMM2: out(4096x1024) = y(4096x2048) @ W_out(1024x2048)^T
    int M = ROWS, N = INPUT_DIM, K = INNER;
    int blocks = (M >> 7) * (N >> 7);           // 256 blocks
    wmma_gemm_bf16<<<blocks, 128, 0, stream>>>(yb, wob, out, M, N, K);
  }
}
